// InteractionBlock_67508295958934
// MI455X (gfx1250) — matrix operand order repaired and checked by reading, not compiled, not run on hardware
//
#include <hip/hip_runtime.h>

typedef __attribute__((ext_vector_type(16))) _Float16 v16h;
typedef __attribute__((ext_vector_type(8)))  float    v8f;

#define N_MSG 144   // 16*(1+3+5) message components per edge/node

__global__ void zero_kernel(float* __restrict__ p, int n) {
  int i = blockIdx.x * 256 + threadIdx.x;
  if (i < n) p[i] = 0.0f;
}

// Repack W2 [16 x 768] into W2rT [48 x 256] f16, column (l,h)-major, K=(k,c),
// folding the 1/sqrt(HIDDEN_DIM)=0.25 edge_z scale.
__global__ void prep_w2_kernel(const float* __restrict__ W2, _Float16* __restrict__ W2rT) {
  int idx = blockIdx.x * 256 + threadIdx.x;
  if (idx >= 48 * 256) return;
  int n = idx >> 8;        // column: l*16 + h
  int K = idx & 255;       // K: k*16 + c
  int l = n >> 4, h = n & 15;
  int k = K >> 4, c = K & 15;
  W2rT[idx] = (_Float16)(W2[k * 768 + l * 256 + c * 16 + h] * 0.25f);
}

__launch_bounds__(256)
__global__ void interaction_kernel(const float* __restrict__ xfeat,
                                   const float* __restrict__ coords,
                                   const int*   __restrict__ eidx,
                                   const float* __restrict__ W1,
                                   const _Float16* __restrict__ W2rT,
                                   float* __restrict__ out,
                                   int n_edges)
{
  __shared__ float W1s[256];
  __shared__ float h_s [8][16][16];
  __shared__ float xs_s[8][16][16];
  __shared__ float Y_s [8][16][12];
  __shared__ int   dst_s[8][16];

  const int tid  = threadIdx.x;
  const int wave = tid >> 5;
  const int lane = tid & 31;
  const int row  = lane & 15;
  const int hi   = lane >> 4;

  W1s[tid] = W1[tid];
  __syncthreads();

  // ---- Phase 1: per-edge scalar pipeline (lanes 0..15 of each wave own one edge) ----
  if (lane < 16) {
    int e = blockIdx.x * 128 + wave * 16 + row;
    const bool valid = (e < n_edges);
    const int ec = valid ? e : 0;
    const int s = eidx[ec];
    const int d = eidx[n_edges + ec];

    float rx = coords[3*s+0] - coords[3*d+0];
    float ry = coords[3*s+1] - coords[3*d+1];
    float rz = coords[3*s+2] - coords[3*d+2];
    float dist = sqrtf(rx*rx + ry*ry + rz*rz);
    float inv  = 1.0f / fmaxf(dist, 1e-12f);
    float ux = rx*inv, uy = ry*inv, uz = rz*inv;

    const float S3    = 1.7320508075688772f;
    const float S5    = 2.23606797749979f;
    const float S15   = 3.872983346207417f;
    const float IS4PI = 0.28209479177387814f;  // 1/sqrt(4*pi)
    float Yv[9];
    Yv[0] = IS4PI;
    Yv[1] = S3 * ux * IS4PI;
    Yv[2] = S3 * uy * IS4PI;
    Yv[3] = S3 * uz * IS4PI;
    Yv[4] = S15 * ux * uz * IS4PI;
    Yv[5] = S15 * ux * uy * IS4PI;
    Yv[6] = S5 * (uy*uy - 0.5f*(ux*ux + uz*uz)) * IS4PI;
    Yv[7] = S15 * uy * uz * IS4PI;
    Yv[8] = 0.5f * S15 * (uz*uz - ux*ux) * IS4PI;

    // Bessel basis: sqrt(2/c)*sin(n*pi*d/c)/d * mask * sqrt(16)
    const float CUT = 6.0f;
    float mask  = (dist > 0.0f && dist < CUT) ? 1.0f : 0.0f;
    float xsafe = (dist > 0.0f) ? dist : 1.0f;
    float amp   = 2.309401076758503f * mask / xsafe;
    float barg  = 3.14159265358979323846f * dist / CUT;
    float eb[16];
    #pragma unroll
    for (int n = 0; n < 16; ++n)
      eb[n] = amp * __sinf(barg * (float)(n + 1));

    // h = SILU_GAIN * silu(eb @ W1 / 4)
    #pragma unroll
    for (int j = 0; j < 16; ++j) {
      float acc = 0.0f;
      #pragma unroll
      for (int n = 0; n < 16; ++n) acc = fmaf(eb[n], W1s[n*16 + j], acc);
      acc *= 0.25f;
      h_s[wave][row][j] = 1.679177f * acc / (1.0f + __expf(-acc));
    }

    const float* xp = xfeat + (size_t)s * 16;
    #pragma unroll
    for (int c = 0; c < 16; ++c) xs_s[wave][row][c] = xp[c];
    dst_s[wave][row] = d;
    #pragma unroll
    for (int m = 0; m < 9; ++m) Y_s[wave][row][m] = valid ? Yv[m] : 0.0f;
  }
  __syncthreads();

  // ---- Phase 2: T[16edges x 48] = (h ⊗ xs)[16 x 256] @ W2r[256 x 48] via WMMA f16 ----
  // A-operand (contiguous k order used here, B uses the same): lane row = M, element j -> K=hi*16+j
  // (k = 2s+hi, c = j). A[e, k*16+c] = h[e][k] * xs[e][c].
  v16h xx;
  #pragma unroll
  for (int j = 0; j < 16; ++j) {
    xx[j] = (_Float16)xs_s[wave][row][j];
  }
  const _Float16* b0p = W2rT + ( 0 + row) * 256 + hi * 16;
  const _Float16* b1p = W2rT + (16 + row) * 256 + hi * 16;
  const _Float16* b2p = W2rT + (32 + row) * 256 + hi * 16;

  v8f acc0 = {}, acc1 = {}, acc2 = {};
  #pragma unroll
  for (int s = 0; s < 8; ++s) {
    _Float16 h0 = (_Float16)h_s[wave][row][2*s + hi];
    v16h hh;
    #pragma unroll
    for (int j = 0; j < 8; ++j) { hh[j] = h0; hh[j + 8] = h0; }
    v16h a = xx * hh;                      // packed f16 products, matches A layout
    v16h b0 = *(const v16h*)(b0p + 32 * s);
    v16h b1 = *(const v16h*)(b1p + 32 * s);
    v16h b2 = *(const v16h*)(b2p + 32 * s);
    acc0 = __builtin_amdgcn_wmma_f32_16x16x32_f16(false, a, false, b0, (short)0, acc0, false, false);
    acc1 = __builtin_amdgcn_wmma_f32_16x16x32_f16(false, a, false, b1, (short)0, acc1, false, false);
    acc2 = __builtin_amdgcn_wmma_f32_16x16x32_f16(false, a, false, b2, (short)0, acc2, false, false);
  }

  // ---- Phase 3: message = T[e,l,h] * Y[e,m] * (1/sqrt(16)) / DEGREE_NORM, scatter to dst ----
  // D layout: lane n = row = h-column; VGPR j -> edge m = hi*8 + j.
  const float SC = 0.015625f;  // 0.25 * (1/16)
  #pragma unroll
  for (int j = 0; j < 8; ++j) {
    int el = hi * 8 + j;
    int nd = dst_s[wave][el];
    float* obase = out + (size_t)nd * N_MSG;
    {   // l = 0 (d=1), idx = h
      float tv = acc0[j] * SC;
      atomicAdd(obase + row, tv * Y_s[wave][el][0]);
    }
    {   // l = 1 (d=3), idx = 16 + h*3 + m
      float tv = acc1[j] * SC;
      #pragma unroll
      for (int m = 0; m < 3; ++m)
        atomicAdd(obase + 16 + row * 3 + m, tv * Y_s[wave][el][1 + m]);
    }
    {   // l = 2 (d=5), idx = 64 + h*5 + m
      float tv = acc2[j] * SC;
      #pragma unroll
      for (int m = 0; m < 5; ++m)
        atomicAdd(obase + 64 + row * 5 + m, tv * Y_s[wave][el][4 + m]);
    }
  }
}

extern "C" void kernel_launch(void* const* d_in, const int* in_sizes, int n_in,
                              void* d_out, int out_size, void* d_ws, size_t ws_size,
                              hipStream_t stream) {
  const float* xfeat  = (const float*)d_in[0];   // atomic_features [N,16]
  const float* coords = (const float*)d_in[1];   // coords [N,3]
  const int*   eidx   = (const int*)d_in[2];     // edge_index [2,E]
  const float* W1     = (const float*)d_in[3];   // [16,16]
  const float* W2     = (const float*)d_in[4];   // [16,768]
  float* out = (float*)d_out;
  const int n_edges = in_sizes[2] / 2;

  _Float16* W2rT = (_Float16*)d_ws;              // 48*256 f16 = 24 KB

  zero_kernel<<<(out_size + 255) / 256, 256, 0, stream>>>(out, out_size);
  prep_w2_kernel<<<(48 * 256 + 255) / 256, 256, 0, stream>>>(W2, W2rT);

  int nblocks = (n_edges + 127) / 128;           // 128 edges per block (8 waves x 16)
  interaction_kernel<<<nblocks, 256, 0, stream>>>(xfeat, coords, eidx, W1, W2rT, out, n_edges);
}
